// WindowAttention_3178275799494
// MI455X (gfx1250) — matrix operand order repaired and checked by reading, not compiled, not run on hardware
//
#include <hip/hip_runtime.h>
#include <hip/hip_bf16.h>

typedef _Float16 v16h __attribute__((ext_vector_type(16)));
typedef _Float16 v8h  __attribute__((ext_vector_type(8)));
typedef _Float16 v4h  __attribute__((ext_vector_type(4)));
typedef float    v8f  __attribute__((ext_vector_type(8)));
typedef float    f4   __attribute__((ext_vector_type(4)));
typedef unsigned int u32x4 __attribute__((ext_vector_type(4)));
typedef int          i32x8 __attribute__((ext_vector_type(8)));
typedef int          i32x4 __attribute__((ext_vector_type(4)));

#define DIM    192
#define N_TOK  49
#define NP     64
#define HEADS  6
#define HD     32
#define C3     576   // 3*DIM
#define QKC    384   // q,k row-major columns (q: 0..191, k: 192..383)
#define B_WIN  8192

// ---------------- LDS layout (halves) ----------------
// xh   [0, 12288)        : [64][192] f16 x, reused as attention output
// qk   [12288, 36864)    : [64][384] f16 (Q | K) row-major [tok][d]
//                          (first 37632 B alias: f32 TDM staging of x)
// vT   [36864, 49152)    : [192][64] f16 V transposed [d][tok]
// psc  [49152, 57344)    : [8 waves][16][64] f16 softmax probabilities
#define LDS_QK_OFF   (NP * DIM)
#define LDS_VT_OFF   (LDS_QK_OFF + NP * QKC)
#define LDS_PSC_OFF  (LDS_VT_OFF + HEADS * HD * NP)
#define LDS_HALVES   (LDS_PSC_OFF + 8 * 16 * 64)

// ---------------- workspace layout (bytes) ----------------
#define WS_PROJ_OFF  (DIM * C3 * 2)                  // 221184
#define WS_BIAS_OFF  (WS_PROJ_OFF + DIM * DIM * 2)   // 294912

// Pack a row-major f32 weight W[K][Nw] into WMMA B-operand tile layout:
// tile (kt, nt) -> 512 halves: [lane][i] = W[kt*32 + (lane/16)*16 + i][nt*16 + lane%16]
__global__ void wa_prep_kernel(const float* __restrict__ qkv_w,
                               const float* __restrict__ proj_w,
                               const float* __restrict__ bias_table,
                               const int*   __restrict__ rel_idx,
                               _Float16* __restrict__ packedQKV,
                               _Float16* __restrict__ packedProj,
                               float* __restrict__ biasHNN) {
    int t = blockIdx.x * blockDim.x + threadIdx.x;
    if (t < DIM * C3) {                       // qkv_w: 6 ktiles x 36 ntiles
        int tile = t >> 9, rem = t & 511;
        int lane = rem >> 4, i = rem & 15;
        int kt = tile / 36, nt = tile % 36;
        int k = kt * 32 + (lane >> 4) * 16 + i;
        int n = nt * 16 + (lane & 15);
        packedQKV[t] = (_Float16)qkv_w[k * C3 + n];
        return;
    }
    int t2 = t - DIM * C3;
    if (t2 < DIM * DIM) {                     // proj_w: 6 ktiles x 12 ntiles
        int tile = t2 >> 9, rem = t2 & 511;
        int lane = rem >> 4, i = rem & 15;
        int kt = tile / 12, nt = tile % 12;
        int k = kt * 32 + (lane >> 4) * 16 + i;
        int n = nt * 16 + (lane & 15);
        packedProj[t2] = (_Float16)proj_w[k * DIM + n];
        return;
    }
    int t3 = t2 - DIM * DIM;
    if (t3 < HEADS * N_TOK * N_TOK) {         // gathered relative-position bias [H][49][49]
        int h = t3 / (N_TOK * N_TOK);
        int r = t3 % (N_TOK * N_TOK);
        biasHNN[t3] = bias_table[rel_idx[r] * HEADS + h];
    }
}

// Load a 16x32 f16 A-operand tile from a row-major LDS row.
// Contiguous layout: lane half kh holds K = {kh*16..kh*16+15} in elems 0..15.
__device__ __forceinline__ v16h load_a_tile(const _Float16* row, int k0, int khalf) {
    v8h lo = *(const v8h*)(row + k0 + khalf * 16);
    v8h hi = *(const v8h*)(row + k0 + khalf * 16 + 8);
    v16h r;
#pragma unroll
    for (int i = 0; i < 8; ++i) { r[i] = lo[i]; r[8 + i] = hi[i]; }
    return r;
}

__global__ __launch_bounds__(256, 2)
void wa_main_kernel(const float* __restrict__ x,
                    const float* __restrict__ qkv_b,
                    const float* __restrict__ proj_b,
                    const _Float16* __restrict__ packedQKV,
                    const _Float16* __restrict__ packedProj,
                    const float* __restrict__ biasHNN,
                    float* __restrict__ out) {
    extern __shared__ _Float16 smem[];
    _Float16* xh = smem;                 // [64][192]
    _Float16* qk = smem + LDS_QK_OFF;    // [64][384]
    _Float16* vT = smem + LDS_VT_OFF;    // [192][64]
    _Float16* psc = smem + LDS_PSC_OFF;  // [8][16][64]

    const int tid   = threadIdx.x;
    const int lane  = tid & 31;
    const int wave  = tid >> 5;
    const int l16   = lane & 15;
    const int khalf = lane >> 4;
    const int win   = blockIdx.x;

    __builtin_prefetch(packedQKV, 0, 1);
    __builtin_prefetch(packedProj, 0, 1);

    // ---- Phase 1: TDM-stage x (f32) into LDS, then convert to f16 ----
    const float* xw = x + (size_t)win * N_TOK * DIM;
    if (wave == 0) {
        unsigned long long gaddr = (unsigned long long)(const void*)xw;
        u32x4 g0;
        g0[0] = 1u;                               // count=1, user descriptor
        g0[1] = (unsigned)(LDS_QK_OFF * 2);       // lds_addr (bytes): staging aliases qk
        g0[2] = (unsigned)(gaddr & 0xFFFFFFFFu);  // global_addr[31:0]
        g0[3] = (unsigned)((gaddr >> 32) & 0x01FFFFFFu) | (2u << 30); // addr[56:32], type=2
        i32x8 g1;
        g1[0] = (int)(2u << 16);                  // data_size = 4 bytes
        g1[1] = (int)((unsigned)DIM << 16);       // tensor_dim0[15:0] = 192
        g1[2] = (int)((unsigned)N_TOK << 16);     // tensor_dim1[15:0] = 49
        g1[3] = (int)((unsigned)DIM << 16);       // tile_dim0 = 192
        g1[4] = N_TOK;                            // tile_dim1 = 49, tile_dim2 = 0
        g1[5] = DIM;                              // tensor_dim0_stride[31:0] = 192
        g1[6] = 0;
        g1[7] = 0;
        i32x4 gz4 = {};
        i32x8 gz8 = {};
        __builtin_amdgcn_tensor_load_to_lds(g0, g1, gz4, gz4, gz8, 0);
        __builtin_amdgcn_s_wait_tensorcnt(0);
    }
    __syncthreads();

    // convert staged f32 -> f16 into xh (vectorized), zero-pad rows 49..63
    {
        const float* xs = (const float*)(smem + LDS_QK_OFF);
        for (int idx = tid * 4; idx < N_TOK * DIM; idx += 256 * 4) {
            f4 v = *(const f4*)(xs + idx);
            v4h h;
#pragma unroll
            for (int i = 0; i < 4; ++i) h[i] = (_Float16)v[i];
            *(v4h*)(xh + idx) = h;
        }
        for (int idx = N_TOK * DIM + tid * 8; idx < NP * DIM; idx += 256 * 8) {
            v8h z = {};
            *(v8h*)(xh + idx) = z;
        }
    }
    __syncthreads();

    // ---- Phase 2: qkv = x @ qkv_w + qkv_b  (64x192 @ 192x576) ----
    // Q,K -> row-major qk[tok][0..383]; V -> transposed vT[d][tok]
    for (int tIdx = wave; tIdx < 4 * 36; tIdx += 8) {
        int mt = tIdx / 36, nt = tIdx % 36;
        v8f acc = {};
        const _Float16* arow = &xh[(mt * 16 + l16) * DIM];
#pragma unroll
        for (int kt = 0; kt < 6; ++kt) {
            v16h a = load_a_tile(arow, kt * 32, khalf);
            v16h b = *(const v16h*)(packedQKV + (((size_t)(kt * 36 + nt)) << 9) + lane * 16);
            acc = __builtin_amdgcn_wmma_f32_16x16x32_f16(false, a, false, b,
                                                         (short)0, acc, false, false);
        }
        if (nt < 24) {                       // Q | K: row-major scalar stores
            int col = nt * 16 + l16;
            float bias = qkv_b[col];
#pragma unroll
            for (int j = 0; j < 8; ++j) {
                int row = mt * 16 + khalf * 8 + j;
                qk[row * QKC + col] = (_Float16)(acc[j] + bias);
            }
        } else {                             // V: transposed, one b128 store per lane
            int colp = (nt - 24) * 16 + l16; // 0..191
            float bias = qkv_b[QKC + colp];
            v8h h;
#pragma unroll
            for (int j = 0; j < 8; ++j) h[j] = (_Float16)(acc[j] + bias);
            *(v8h*)&vT[colp * NP + mt * 16 + khalf * 8] = h;
        }
    }
    __syncthreads();

    // ---- Phase 3: per (head, m-tile): scores -> softmax -> @ V ----
    const float scale = 0.17677669529663687f;   // 1/sqrt(32)
    for (int it = 0; it < 3; ++it) {
        int rb   = wave + 8 * it;               // 0..23
        int head = rb >> 2;
        int mt   = rb & 3;

        // scores S[16][64] = Q_tile (16x32) @ K^T tiles (32x16)
        v16h aq = load_a_tile(&qk[(mt * 16 + l16) * QKC], head * HD, khalf);
        v8f s[4];
#pragma unroll
        for (int nt2 = 0; nt2 < 4; ++nt2) {
            v16h b = *(const v16h*)&qk[(nt2 * 16 + l16) * QKC + DIM + head * HD + khalf * 16];
            v8f z = {};
            s[nt2] = __builtin_amdgcn_wmma_f32_16x16x32_f16(false, aq, false, b,
                                                            (short)0, z, false, false);
        }
        // scale + relative-position bias, mask padding
#pragma unroll
        for (int nt2 = 0; nt2 < 4; ++nt2) {
            int col = nt2 * 16 + l16;
#pragma unroll
            for (int j = 0; j < 8; ++j) {
                int row = mt * 16 + khalf * 8 + j;
                float v = -1e30f;
                if (col < N_TOK && row < N_TOK)
                    v = s[nt2][j] * scale + biasHNN[(head * N_TOK + row) * N_TOK + col];
                s[nt2][j] = v;
            }
        }
        // row-wise softmax: each row lives in one 16-lane half-wave
        float rsum[8];
#pragma unroll
        for (int j = 0; j < 8; ++j) {
            float m = fmaxf(fmaxf(s[0][j], s[1][j]), fmaxf(s[2][j], s[3][j]));
            for (int off = 1; off < 16; off <<= 1)
                m = fmaxf(m, __shfl_xor(m, off, 16));
#pragma unroll
            for (int nt2 = 0; nt2 < 4; ++nt2)
                s[nt2][j] = __expf(s[nt2][j] - m);
            float t = s[0][j] + s[1][j] + s[2][j] + s[3][j];
            for (int off = 1; off < 16; off <<= 1)
                t += __shfl_xor(t, off, 16);
            rsum[j] = t;
        }
        // P -> per-wave LDS scratch (f16), row-major [16][64]
        _Float16* pw = psc + wave * (16 * 64);
#pragma unroll
        for (int nt2 = 0; nt2 < 4; ++nt2) {
            int col = nt2 * 16 + l16;
#pragma unroll
            for (int j = 0; j < 8; ++j)
                pw[(khalf * 8 + j) * 64 + col] = (_Float16)(s[nt2][j] / rsum[j]);
        }
        __syncthreads();

        // out_tile (16x32) = P (16x64) @ V (64x32); V read from vT with b128 loads
        v8f o0 = {}, o1 = {};
#pragma unroll
        for (int k0 = 0; k0 < 64; k0 += 32) {
            v16h a  = load_a_tile(&pw[l16 * 64], k0, khalf);
            v16h b0 = *(const v16h*)&vT[(head * HD + l16) * NP + k0 + khalf * 16];
            v16h b1 = *(const v16h*)&vT[(head * HD + 16 + l16) * NP + k0 + khalf * 16];
            o0 = __builtin_amdgcn_wmma_f32_16x16x32_f16(false, a, false, b0,
                                                        (short)0, o0, false, false);
            o1 = __builtin_amdgcn_wmma_f32_16x16x32_f16(false, a, false, b1,
                                                        (short)0, o1, false, false);
        }
        // store head output (f16) into xh (x no longer needed)
#pragma unroll
        for (int j = 0; j < 8; ++j) {
            int row = mt * 16 + khalf * 8 + j;
            xh[row * DIM + head * HD + l16]      = (_Float16)o0[j];
            xh[row * DIM + head * HD + 16 + l16] = (_Float16)o1[j];
        }
    }
    __syncthreads();

    // ---- Phase 4: out = attn_out @ proj_w + proj_b  (64x192 @ 192x192) ----
    float* outw = out + (size_t)win * N_TOK * DIM;
    for (int tIdx = wave; tIdx < 4 * 12; tIdx += 8) {
        int mt = tIdx / 12, nt = tIdx % 12;
        v8f acc = {};
        const _Float16* arow = &xh[(mt * 16 + l16) * DIM];
#pragma unroll
        for (int kt = 0; kt < 6; ++kt) {
            v16h a = load_a_tile(arow, kt * 32, khalf);
            v16h b = *(const v16h*)(packedProj + (((size_t)(kt * 12 + nt)) << 9) + lane * 16);
            acc = __builtin_amdgcn_wmma_f32_16x16x32_f16(false, a, false, b,
                                                         (short)0, acc, false, false);
        }
        int col = nt * 16 + l16;
        float bias = proj_b[col];
#pragma unroll
        for (int j = 0; j < 8; ++j) {
            int row = mt * 16 + khalf * 8 + j;
            if (row < N_TOK)
                outw[row * DIM + col] = acc[j] + bias;
        }
    }
}

extern "C" void kernel_launch(void* const* d_in, const int* in_sizes, int n_in,
                              void* d_out, int out_size, void* d_ws, size_t ws_size,
                              hipStream_t stream) {
    const float* x          = (const float*)d_in[0];
    const float* qkv_w      = (const float*)d_in[1];
    const float* qkv_b      = (const float*)d_in[2];
    const float* proj_w     = (const float*)d_in[3];
    const float* proj_b     = (const float*)d_in[4];
    const float* bias_table = (const float*)d_in[5];
    const int*   rel_idx    = (const int*)d_in[6];

    char* ws = (char*)d_ws;
    _Float16* packedQKV  = (_Float16*)ws;
    _Float16* packedProj = (_Float16*)(ws + WS_PROJ_OFF);
    float*    biasHNN    = (float*)(ws + WS_BIAS_OFF);

    int prepTot = DIM * C3 + DIM * DIM + HEADS * N_TOK * N_TOK;
    wa_prep_kernel<<<(prepTot + 255) / 256, 256, 0, stream>>>(
        qkv_w, proj_w, bias_table, rel_idx, packedQKV, packedProj, biasHNN);

    size_t lds_bytes = (size_t)LDS_HALVES * sizeof(_Float16); // 112 KB
    wa_main_kernel<<<B_WIN, 256, lds_bytes, stream>>>(
        x, qkv_b, proj_b, packedQKV, packedProj, biasHNN, (float*)d_out);
}
